// DSConv_pro_2430951489993
// MI455X (gfx1250) — matrix operand order repaired and checked by reading, not compiled, not run on hardware
//
#include <hip/hip_runtime.h>

// ---------------------------------------------------------------------------
// DSConv "snake" deformable conv for MI455X (gfx1250, wave32, WMMA).
//
// Roofline: ~6 GFLOP total, working set < 40MB (fits 192MB L2) -> gather/L2
// bound. Strategy: never materialize the 150MB "deformed" tensor; fuse the
// 2-tap y-interpolation gather with the 576-deep DSC reduction using
// v_wmma_f32_16x16x32_f16 (f32 accumulate) on LDS-staged 32-pixel tiles.
// x-offsets are integers => bilinear degenerates to 2 loads/sample.
// Only 10 of the 18 offset-conv channels are ever consumed => compute 10.
// ---------------------------------------------------------------------------

typedef __attribute__((ext_vector_type(16))) _Float16 v16h;
typedef __attribute__((ext_vector_type(8)))  _Float16 v8h;
typedef __attribute__((ext_vector_type(8)))  float    v8f;

#define Bn     4
#define CIN    64
#define COUT   64
#define Hh     128
#define Ww     128
#define HW     (Hh*Ww)          // 16384
#define Kk     9
#define COFF   10               // offset-conv channels actually needed
#define KFLAT  (Kk*CIN)         // 576
#define NCHUNK (KFLAT/32)       // 18 WMMA K-chunks
#define EPS    1e-5f

// workspace layout (float offsets)
#define WS_OFF    0                            // B*COFF*HW   = 655360
#define WS_STATS  (WS_OFF + Bn*COFF*HW)        // 64 (uses 40: mu,rsig x B*5)
#define WS_Y0     (WS_STATS + 64)              // B*9*HW ints = 589824
#define WS_WY     (WS_Y0 + Bn*Kk*HW)           // B*9*HW floats
#define WS_STATS2 (WS_WY + Bn*Kk*HW)           // 128 (uses 128: B*16*2)
#define WS_WPACK  (WS_STATS2 + 128)            // 36864 halfs (18432 floats)

// ---------------------------------------------------------------------------
// K0: repack w_dsc (Cout,Cin,9,1) f32 into per-lane WMMA A-fragment order
// (16-bit A 16x32 layout, ISA 7.12.2), flat K index = k*64 + ci.
// wpack[((cot*18+chunk)*32+lane)*16 + v*2+h]
// ---------------------------------------------------------------------------
__global__ void k0_pack_w(const float* __restrict__ w_dsc,
                          _Float16* __restrict__ wp) {
  int e = blockIdx.x*256 + threadIdx.x;            // 36864 elements
  int hh   =  e        & 1;
  int v    = (e >> 1)  & 7;
  int lane = (e >> 4)  & 31;
  int chunk= (e >> 9)  % NCHUNK;
  int cot  =  e / (512*NCHUNK);
  // A 16x32 f16 layout: lanes0-15: V0..3 -> K0..7, V4..7 -> K16..23;
  //                     lanes16-31: +8
  int kl = 2*v + (v >= 4 ? 8 : 0) + (lane >= 16 ? 8 : 0) + hh;
  int kf = chunk*32 + kl;                          // flat K in [0,576)
  int ci = kf & 63;
  int kk = kf >> 6;
  int co = cot*16 + (lane & 15);
  wp[e] = (_Float16)w_dsc[(co*CIN + ci)*Kk + kk];
}

// ---------------------------------------------------------------------------
// K1: 3x3 SAME conv 64 -> 10 channels (only channels 0..9 feed the rest).
// One thread per (b,h,w); weights staged in LDS (23KB), broadcast reads.
// ---------------------------------------------------------------------------
__global__ void k1_offset_conv(const float* __restrict__ x,
                               const float* __restrict__ w_off,
                               const float* __restrict__ b_off,
                               float* __restrict__ off) {
  __shared__ float wl[COFF*CIN*9];
  for (int i = threadIdx.x; i < COFF*CIN*9; i += 256) wl[i] = w_off[i];
  __syncthreads();
  int tid = blockIdx.x*256 + threadIdx.x;          // 65536
  int b = tid / HW, p = tid % HW;
  int h = p / Ww,  w = p % Ww;
  float acc[COFF];
#pragma unroll
  for (int c = 0; c < COFF; ++c) acc[c] = b_off[c];
  const float* xb = x + b*CIN*HW;
  for (int ky = 0; ky < 3; ++ky) {
    int hy = h + ky - 1;
    if (hy < 0 || hy >= Hh) continue;
    for (int kx = 0; kx < 3; ++kx) {
      int wx = w + kx - 1;
      if (wx < 0 || wx >= Ww) continue;
      int j = ky*3 + kx;
      for (int ci = 0; ci < CIN; ++ci) {
        float xv = xb[ci*HW + hy*Ww + wx];
#pragma unroll
        for (int c = 0; c < COFF; ++c)
          acc[c] = fmaf(xv, wl[c*CIN*9 + ci*9 + j], acc[c]);
      }
    }
  }
  for (int c = 0; c < COFF; ++c) off[(b*COFF + c)*HW + p] = acc[c];
}

// ---------------------------------------------------------------------------
// K2: GroupNorm stats for offset conv, groups of 2 contiguous channels.
// Block per (b, g<5). Stores mu, rsig.
// ---------------------------------------------------------------------------
__global__ void k2_gn_off_stats(const float* __restrict__ off,
                                float* __restrict__ stats) {
  int bg = blockIdx.x;                              // 0..19
  int b = bg / 5, g = bg % 5;
  const float* base = off + (b*COFF + 2*g)*HW;      // 2 contiguous channels
  float s = 0.f, sq = 0.f;
  for (int i = threadIdx.x; i < 2*HW; i += 256) {
    float v = base[i]; s += v; sq += v*v;
  }
  __shared__ float rs[256], rq[256];
  rs[threadIdx.x] = s; rq[threadIdx.x] = sq;
  __syncthreads();
  for (int st = 128; st > 0; st >>= 1) {
    if (threadIdx.x < st) {
      rs[threadIdx.x] += rs[threadIdx.x + st];
      rq[threadIdx.x] += rq[threadIdx.x + st];
    }
    __syncthreads();
  }
  if (threadIdx.x == 0) {
    float n   = (float)(2*HW);
    float mu  = rs[0] / n;
    float var = rq[0] / n - mu*mu;
    stats[bg*2]     = mu;
    stats[bg*2 + 1] = rsqrtf(var + EPS);
  }
}

// ---------------------------------------------------------------------------
// K3: normalize + tanh + snake cumsum -> per-(b,k,pixel) y0 (int) & wy tables.
// x-offsets are integer (w + k - 4), handled at sample time.
// ---------------------------------------------------------------------------
__global__ void k3_make_offsets(const float* __restrict__ off,
                                const float* __restrict__ stats,
                                const float* __restrict__ gamma,
                                const float* __restrict__ beta,
                                int*   __restrict__ y0o,
                                float* __restrict__ wyo) {
  int tid = blockIdx.x*256 + threadIdx.x;          // 65536
  int b = tid / HW, p = tid % HW;
  int h = p / Ww;
  float t[9];
#pragma unroll
  for (int k = 0; k < 9; ++k) {
    int g = k >> 1;                                // channel k in group k/2
    float mu = stats[(b*5 + g)*2];
    float rsg= stats[(b*5 + g)*2 + 1];
    float v  = (off[(b*COFF + k)*HW + p] - mu) * rsg * gamma[k] + beta[k];
    t[k] = tanhf(v);
  }
  float nw[9];
  nw[4] = 0.f;
  nw[5] = t[5]; nw[6] = nw[5] + t[6]; nw[7] = nw[6] + t[7]; nw[8] = nw[7] + t[8];
  nw[3] = t[3]; nw[2] = nw[3] + t[2]; nw[1] = nw[2] + t[1]; nw[0] = nw[1] + t[0];
#pragma unroll
  for (int k = 0; k < 9; ++k) {
    float yc = fminf(fmaxf((float)h + nw[k], 0.f), (float)(Hh - 1)); // EXT=1
    float y0 = floorf(yc);
    int idx  = (b*9 + k)*HW + p;
    y0o[idx] = (int)y0;
    wyo[idx] = yc - y0;
  }
}

// ---------------------------------------------------------------------------
// K4: fused gather + DSC conv via WMMA.
// Block = 256 thr = 8 waves, tile = 64 co x 32 pixels, Kdim = 576.
// Stage sampled f16 tile in LDS (32 x 584 halfs, padded stride), 1 barrier,
// then each wave (cot 0..3, ptile 0..1) runs 18x v_wmma_f32_16x16x32_f16.
// ---------------------------------------------------------------------------
#define LDST 584   // 576 + pad; keeps 16B alignment, breaks bank conflicts
__global__ void k4_sample_wmma(const float* __restrict__ x,
                               const int*   __restrict__ y0t,
                               const float* __restrict__ wyt,
                               const _Float16* __restrict__ wp,
                               const float* __restrict__ b_dsc,
                               float* __restrict__ out) {
  __shared__ _Float16 smem[32*LDST];
  int blk   = blockIdx.x;                          // 2048 = B * 512
  int b     = blk >> 9;
  int pbase = (blk & 511) * 32;
  int t     = threadIdx.x;

  // --- sampling: 32 pixels x 9 k x 64 ci, 2-tap y-interp, x integer ---
  {
    int pit = t & 31;                              // pixel in tile
    int cig = t >> 5;                              // 8 ci per thread
    int p   = pbase + pit;
    int w   = p & 127;
    const float* xb = x + b*CIN*HW;
#pragma unroll
    for (int k = 0; k < 9; ++k) {
      int   idx = (b*9 + k)*HW + p;
      int   y0  = y0t[idx];
      float wy  = wyt[idx];
      int   y1  = min(y0 + 1, Hh - 1);
      int   xi  = min(max(w + k - 4, 0), Ww - 1);
      int   o0  = y0*Ww + xi, o1 = y1*Ww + xi;
#pragma unroll
      for (int c8 = 0; c8 < 8; ++c8) {
        int ci = cig*8 + c8;
        const float* xc = xb + ci*HW;
        float v0 = xc[o0], v1 = xc[o1];
        smem[pit*LDST + k*64 + ci] = (_Float16)(v0 + wy*(v1 - v0));
      }
    }
  }
  __syncthreads();

  // --- WMMA: out[16co x 16pix] += A[16x32] * B[32x16] over 18 chunks ---
  int lane  = t & 31;
  int wv    = t >> 5;
  int cot   = wv & 3;
  int ptile = wv >> 2;
  int nn    = lane & 15;
  int khalf = lane >> 4;                           // B: lanes>=16 hold K8..15,K24..31
  v8f acc = {};
  const _Float16* wbase = wp + (cot*NCHUNK)*32*16 + lane*16;
  const _Float16* bbase = &smem[(ptile*16 + nn)*LDST + khalf*8];
#pragma unroll
  for (int ch = 0; ch < NCHUNK; ++ch) {
    v16h a   = *(const v16h*)(wbase + ch*32*16);   // 32B contiguous, aligned
    v8h  blo = *(const v8h*)(bbase + ch*32);
    v8h  bhi = *(const v8h*)(bbase + ch*32 + 16);
    v16h bf;
#pragma unroll
    for (int i = 0; i < 8; ++i) { bf[i] = blo[i]; bf[i+8] = bhi[i]; }
    acc = __builtin_amdgcn_wmma_f32_16x16x32_f16(
        /*neg_a=*/false, a, /*neg_b=*/false, bf,
        /*c_mod=*/(short)0, acc, /*reuse_a=*/false, /*reuse_b=*/false);
  }

  // --- store raw conv output + bias (C/D layout: M = r + 8*(lane>=16)) ---
  int pix  = pbase + ptile*16 + nn;
  int mrow = (lane >> 4) * 8;
#pragma unroll
  for (int r = 0; r < 8; ++r) {
    int co = cot*16 + mrow + r;
    out[(b*COUT + co)*HW + pix] = acc[r] + b_dsc[co];
  }
}

// ---------------------------------------------------------------------------
// K5: final GroupNorm stats (16 groups of 4 contiguous channels).
// ---------------------------------------------------------------------------
__global__ void k5_gn_out_stats(const float* __restrict__ out,
                                float* __restrict__ stats2) {
  int bg = blockIdx.x;                             // 0..63
  int b = bg >> 4, g = bg & 15;
  const float* base = out + (b*COUT + g*4)*HW;     // 4 contiguous channels
  float s = 0.f, sq = 0.f;
  for (int i = threadIdx.x; i < 4*HW; i += 256) {
    float v = base[i]; s += v; sq += v*v;
  }
  __shared__ float rs[256], rq[256];
  rs[threadIdx.x] = s; rq[threadIdx.x] = sq;
  __syncthreads();
  for (int st = 128; st > 0; st >>= 1) {
    if (threadIdx.x < st) {
      rs[threadIdx.x] += rs[threadIdx.x + st];
      rq[threadIdx.x] += rq[threadIdx.x + st];
    }
    __syncthreads();
  }
  if (threadIdx.x == 0) {
    float n   = (float)(4*HW);
    float mu  = rs[0] / n;
    float var = rq[0] / n - mu*mu;
    stats2[bg*2]     = mu;
    stats2[bg*2 + 1] = rsqrtf(var + EPS);
  }
}

// ---------------------------------------------------------------------------
// K6: in-place normalize + ReLU on d_out.
// ---------------------------------------------------------------------------
__global__ void k6_gn_relu(float* __restrict__ out,
                           const float* __restrict__ stats2,
                           const float* __restrict__ gamma,
                           const float* __restrict__ beta) {
  int idx = blockIdx.x*256 + threadIdx.x;          // 4,194,304
  int c = (idx >> 14) & 63;
  int b =  idx >> 20;
  int g =  c >> 2;
  float mu = stats2[(b*16 + g)*2];
  float rsg= stats2[(b*16 + g)*2 + 1];
  float v  = (out[idx] - mu) * rsg * gamma[c] + beta[c];
  out[idx] = fmaxf(v, 0.f);
}

// ---------------------------------------------------------------------------
extern "C" void kernel_launch(void* const* d_in, const int* in_sizes, int n_in,
                              void* d_out, int out_size, void* d_ws, size_t ws_size,
                              hipStream_t stream) {
  const float* x        = (const float*)d_in[0];
  const float* w_off    = (const float*)d_in[1];
  const float* b_off    = (const float*)d_in[2];
  const float* g_gn_off = (const float*)d_in[3];
  const float* b_gn_off = (const float*)d_in[4];
  const float* w_dsc    = (const float*)d_in[5];
  const float* b_dsc    = (const float*)d_in[6];
  const float* g_gn     = (const float*)d_in[7];
  const float* b_gn     = (const float*)d_in[8];
  float* out = (float*)d_out;
  float* ws  = (float*)d_ws;

  float*    off    = ws + WS_OFF;
  float*    stats  = ws + WS_STATS;
  int*      y0t    = (int*)(ws + WS_Y0);
  float*    wyt    = ws + WS_WY;
  float*    stats2 = ws + WS_STATS2;
  _Float16* wp     = (_Float16*)(ws + WS_WPACK);

  k0_pack_w      <<<144,   256, 0, stream>>>(w_dsc, wp);
  k1_offset_conv <<<256,   256, 0, stream>>>(x, w_off, b_off, off);
  k2_gn_off_stats<<<20,    256, 0, stream>>>(off, stats);
  k3_make_offsets<<<256,   256, 0, stream>>>(off, stats, g_gn_off, b_gn_off, y0t, wyt);
  k4_sample_wmma <<<2048,  256, 0, stream>>>(x, y0t, wyt, wp, b_dsc, out);
  k5_gn_out_stats<<<64,    256, 0, stream>>>(out, stats2);
  k6_gn_relu     <<<16384, 256, 0, stream>>>(out, stats2, g_gn, b_gn);
}
